// GraphAttentionLayer_91182155694852
// MI455X (gfx1250) — matrix operand order repaired and checked by reading, not compiled, not run on hardware
//
#include <hip/hip_runtime.h>
#include <hip/hip_bf16.h>

// ---------------------------------------------------------------------------
// GAT dense attention on MI455X (gfx1250, wave32, WMMA).
//   h  = x @ W            (8192x512, K=128)      -> v_wmma_f32_16x16x32_f16
//   s  = h . a_src, d = h . a_dst (per row)
//   e  = leaky(s_i + d_j); softmax_j; out = attn @ h (streamed, never
//        materializing the 256MB e tensor)       -> v_wmma_f32_16x16x32_f16
// Row-max trick: leaky is monotone => max_j leaky(s_i+d_j) = leaky(s_i+dmax).
// CDNA5 paths: v_wmma_f32_16x16x32_f16, global_load_async_to_lds_b128 +
// s_wait_asynccnt double-buffered staging.
// ---------------------------------------------------------------------------

typedef __attribute__((ext_vector_type(16))) _Float16 v16h;
typedef __attribute__((ext_vector_type(8)))  float    v8f;

#define GAT_ALPHA 0.2f

__device__ __forceinline__ float leaky(float e) { return e >= 0.f ? e : GAT_ALPHA * e; }

// ---------------------------------------------------------------------------
// Kernel 1: h = x @ W, written TRANSPOSED as hT[bh][d][n]  (f16, 8 MB)
//   x: (8*1024, 128) f32 row-major, W: (128, 512) f32 row-major.
//   One 16x16 output tile per wave. All 8 waves of a block share n0, so the
//   128x16 W slab is staged once into LDS (f16, k-major, stride 136).
// ---------------------------------------------------------------------------
__global__ __launch_bounds__(256) void k_gemm_xw(const float* __restrict__ x,
                                                 const float* __restrict__ W,
                                                 _Float16* __restrict__ hT)
{
    __shared__ _Float16 WlT[16 * 136];          // [col][k], padded rows (272B)

    const int tid  = threadIdx.x;
    const int wave = tid >> 5;
    const int lane = tid & 31;
    const int half = lane >> 4;
    const int lm   = lane & 15;

    const int nT = blockIdx.x >> 6;             // same for all 8 waves
    const int mT = (blockIdx.x & 63) * 8 + wave;
    const int n0 = nT * 16;
    const int m0 = mT * 16;

    // ---- stage W[0:128, n0:n0+16] -> WlT[col][k] (f16), one-time
    {
        const int k    = tid >> 1;              // 0..127
        const int cgrp = (tid & 1) * 8;         // 0 or 8
        const float4 w0 = *(const float4*)(W + (size_t)k * 512 + n0 + cgrp);
        const float4 w1 = *(const float4*)(W + (size_t)k * 512 + n0 + cgrp + 4);
        WlT[(cgrp + 0) * 136 + k] = (_Float16)w0.x;
        WlT[(cgrp + 1) * 136 + k] = (_Float16)w0.y;
        WlT[(cgrp + 2) * 136 + k] = (_Float16)w0.z;
        WlT[(cgrp + 3) * 136 + k] = (_Float16)w0.w;
        WlT[(cgrp + 4) * 136 + k] = (_Float16)w1.x;
        WlT[(cgrp + 5) * 136 + k] = (_Float16)w1.y;
        WlT[(cgrp + 6) * 136 + k] = (_Float16)w1.z;
        WlT[(cgrp + 7) * 136 + k] = (_Float16)w1.w;
    }
    __syncthreads();

    const int m = m0 + lm;                      // this lane's A row
    const float* __restrict__ xrow = x + (size_t)m * 128;

    v8f acc = {};
    #pragma unroll
    for (int k0 = 0; k0 < 128; k0 += 32) {
        // ---- A fragment (16x32 f16): lane holds K = kb+{0..7} and 16+kb+{0..7}
        v16h A;
        {
            const int kb = k0 + half * 8;
            const float4 r0 = *(const float4*)(xrow + kb);
            const float4 r1 = *(const float4*)(xrow + kb + 4);
            const float4 r2 = *(const float4*)(xrow + kb + 16);
            const float4 r3 = *(const float4*)(xrow + kb + 20);
            A[0]=(_Float16)r0.x; A[1]=(_Float16)r0.y; A[2]=(_Float16)r0.z; A[3]=(_Float16)r0.w;
            A[4]=(_Float16)r1.x; A[5]=(_Float16)r1.y; A[6]=(_Float16)r1.z; A[7]=(_Float16)r1.w;
            A[8]=(_Float16)r2.x; A[9]=(_Float16)r2.y; A[10]=(_Float16)r2.z; A[11]=(_Float16)r2.w;
            A[12]=(_Float16)r3.x; A[13]=(_Float16)r3.y; A[14]=(_Float16)r3.z; A[15]=(_Float16)r3.w;
        }
        // ---- B fragment (32x16 f16): lane holds column lm, K = kb+{0..7} and 16+kb+{0..7}
        v16h Bf;
        {
            const _Float16* bp = &WlT[lm * 136 + k0 + half * 8];
            union { v16h v; uint4 u[2]; } t;
            t.u[0] = *(const uint4*)bp;
            t.u[1] = *(const uint4*)(bp + 16);
            Bf = t.v;
        }
        acc = __builtin_amdgcn_wmma_f32_16x16x32_f16(false, A, false, Bf,
                                                     (short)0, acc, false, false);
    }

    // ---- C/D: lane holds column n0+lm, rows m0 + 8*half + v. Scatter -> hT.
    const int n  = n0 + lm;
    const int hh = n >> 6;        // head
    const int dd = n & 63;        // dim within head
    #pragma unroll
    for (int v = 0; v < 8; ++v) {
        const int row = m0 + 8 * half + v;
        const int seq = row & 1023;
        const int b   = row >> 10;
        hT[((size_t)((b * 8 + hh) * 64 + dd)) * 1024 + seq] = (_Float16)acc[v];
    }
}

// ---------------------------------------------------------------------------
// Kernel 2: s[bh][n] = h . a_src ; d[bh][n] = h . a_dst   (coalesced over n)
// ---------------------------------------------------------------------------
__global__ __launch_bounds__(256) void k_sd(const _Float16* __restrict__ hT,
                                            const float* __restrict__ a,
                                            float* __restrict__ sArr,
                                            float* __restrict__ dArr)
{
    const int bh = blockIdx.x >> 2;
    const int n  = ((blockIdx.x & 3) << 8) + threadIdx.x;
    const _Float16* __restrict__ base = hT + (size_t)bh * 64 * 1024 + n;
    float s = 0.f, d = 0.f;
    #pragma unroll 8
    for (int dd = 0; dd < 64; ++dd) {
        const float hv = (float)base[(size_t)dd * 1024];
        s += hv * a[dd];
        d += hv * a[64 + dd];
    }
    sArr[bh * 1024 + n] = s;
    dArr[bh * 1024 + n] = d;
}

// ---------------------------------------------------------------------------
// Kernel 3: streaming softmax-attention.  Grid: 256 WGs = (bh, 256-row chunk).
// 8 waves x 2 i-tiles(16 rows) each; j streamed in 32-wide tiles, staged via
// double-buffered global_load_async_to_lds_b128 (ASYNCcnt pipeline).
// ---------------------------------------------------------------------------
__global__ __launch_bounds__(256) void k_attn(const _Float16* __restrict__ hT,
                                              const float* __restrict__ sArr,
                                              const float* __restrict__ dArr,
                                              float* __restrict__ out)
{
    __shared__ float    d_lds[1024];        // all d_j for this (b,h)
    __shared__ float    red[256];
    __shared__ float    denomLds[256];      // per-row softmax denominators
    __shared__ _Float16 bt[2][64 * 40];     // 2 x (64 d-rows x 32 j + pad) f16

    const int bh    = blockIdx.x >> 2;
    const int chunk = blockIdx.x & 3;
    const int tid   = threadIdx.x;
    const int wave  = tid >> 5;
    const int lane  = tid & 31;
    const int half  = lane >> 4;
    const int lm    = lane & 15;

    const _Float16* __restrict__ hbase = hT + (size_t)bh * 64 * 1024;
    const int drow = tid >> 2;              // staging: this thread's d-row
    const int doff = (tid & 3) * 8;         // and 8-element j-offset

    // Async stage of one 32x64 value tile into LDS buffer `buf` (16B/lane).
    // dsaddr = LDS offset (low 32 bits of the generic pointer), GV addressing.
    auto stage = [&](int jt, int buf) {
        const unsigned int lds =
            (unsigned int)(unsigned long long)&bt[buf][drow * 40 + doff];
        const unsigned long long ga =
            (unsigned long long)(hbase + (size_t)drow * 1024 + jt + doff);
        asm volatile("global_load_async_to_lds_b128 %0, %1, off"
                     :: "v"(lds), "v"(ga) : "memory");
    };

    stage(0, 0);                            // prefetch tile 0 (overlaps reduce)

    // ---- load d[], block-reduce dmax (async copy of tile 0 runs underneath)
    float lmax = -1e30f;
    #pragma unroll
    for (int k = 0; k < 4; ++k) {
        const float v = dArr[bh * 1024 + k * 256 + tid];
        d_lds[k * 256 + tid] = v;
        lmax = fmaxf(lmax, v);
    }
    red[tid] = lmax;
    __syncthreads();
    for (int off = 128; off > 0; off >>= 1) {
        if (tid < off) red[tid] = fmaxf(red[tid], red[tid + off]);
        __syncthreads();
    }
    const float dmax = red[0];

    // ---- per-lane row state (2 i-tiles of 16 rows per wave)
    const int iLocalBase = wave * 32;       // within 256-row chunk
    float sI[2], mI[2], denomAcc[2] = {0.f, 0.f};
    #pragma unroll
    for (int it = 0; it < 2; ++it) {
        const int row = chunk * 256 + iLocalBase + it * 16 + lm;
        const float sv = sArr[bh * 1024 + row];
        sI[it] = sv;
        mI[it] = leaky(sv + dmax);          // exact row max of logits
    }

    v8f acc[2][4];
    #pragma unroll
    for (int it = 0; it < 2; ++it)
        #pragma unroll
        for (int dt = 0; dt < 4; ++dt) acc[it][dt] = (v8f){};

    for (int jt = 0; jt < 1024; jt += 32) {
        const int cur = (jt >> 5) & 1;
        // my tile-`jt` async op has landed in LDS:
        asm volatile("s_wait_asynccnt 0" ::: "memory");
        // everyone's tile has landed; also fences last iteration's bt reads:
        __syncthreads();
        if (jt + 32 < 1024) stage(jt + 32, cur ^ 1);   // DMA next tile now

        // ---- B fragments: lane holds column d = dt*16+lm, K = 8*half+{0..7} and 16+8*half+{0..7}
        v16h Bf[4];
        #pragma unroll
        for (int dt = 0; dt < 4; ++dt) {
            const _Float16* bp = &bt[cur][(dt * 16 + lm) * 40 + half * 8];
            union { v16h v; uint4 u[2]; } t;
            t.u[0] = *(const uint4*)bp;
            t.u[1] = *(const uint4*)(bp + 16);
            Bf[dt] = t.v;
        }

        // ---- A fragment = P tile (probabilities), built in-register
        #pragma unroll
        for (int it = 0; it < 2; ++it) {
            v16h A;
            float dacc = 0.f;
            const int kb = half * 8;
            #pragma unroll
            for (int t = 0; t < 8; ++t) {
                const float p1 = __expf(leaky(sI[it] + d_lds[jt + kb + t])      - mI[it]);
                const float p2 = __expf(leaky(sI[it] + d_lds[jt + 16 + kb + t]) - mI[it]);
                dacc += p1 + p2;
                A[t]     = (_Float16)p1;
                A[8 + t] = (_Float16)p2;
            }
            denomAcc[it] += dacc;
            #pragma unroll
            for (int dt = 0; dt < 4; ++dt)
                acc[it][dt] = __builtin_amdgcn_wmma_f32_16x16x32_f16(
                    false, A, false, Bf[dt], (short)0, acc[it][dt], false, false);
        }
    }

    // ---- finalize denominators: lane L covers half the j's of row lm; pair
    // lane (L xor 16) covers the other half (wave32 shuffle).
    #pragma unroll
    for (int it = 0; it < 2; ++it) {
        const float tot = denomAcc[it] + __shfl_xor(denomAcc[it], 16, 32);
        if (half == 0) denomLds[iLocalBase + it * 16 + lm] = tot;
    }
    __syncthreads();

    // ---- writeout: lane holds column dcol, rows iLocalBase+it*16+8*half+v
    const int b  = bh >> 3;
    const int hh = bh & 7;
    #pragma unroll
    for (int it = 0; it < 2; ++it) {
        #pragma unroll
        for (int dt = 0; dt < 4; ++dt) {
            const int dcol = dt * 16 + lm;
            #pragma unroll
            for (int v = 0; v < 8; ++v) {
                const int il  = iLocalBase + it * 16 + 8 * half + v;
                const int seq = chunk * 256 + il;
                const float r = acc[it][dt][v] / denomLds[il];
                out[((size_t)(b * 1024 + seq)) * 512 + hh * 64 + dcol] = r;
            }
        }
    }
}

// ---------------------------------------------------------------------------
extern "C" void kernel_launch(void* const* d_in, const int* in_sizes, int n_in,
                              void* d_out, int out_size, void* d_ws, size_t ws_size,
                              hipStream_t stream)
{
    const float* x = (const float*)d_in[0];   // (8,1024,128) f32
    const float* W = (const float*)d_in[1];   // (128,512)    f32
    const float* a = (const float*)d_in[2];   // (128,)       f32
    float* out = (float*)d_out;               // (8,1024,512) f32

    // scratch: hT (64*64*1024 f16 = 8 MB) | s (64*1024 f32) | d (64*1024 f32)
    _Float16* hT  = (_Float16*)d_ws;
    float* sArr   = (float*)((char*)d_ws + (size_t)64 * 64 * 1024 * sizeof(_Float16));
    float* dArr   = sArr + 64 * 1024;

    k_gemm_xw<<<2048, 256, 0, stream>>>(x, W, hT);
    k_sd    <<<256,  256, 0, stream>>>(hT, a, sArr, dArr);
    k_attn  <<<256,  256, 0, stream>>>(hT, sArr, dArr, out);
}
